// Decoder_52896817217569
// MI455X (gfx1250) — matrix operand order repaired and checked by reading, not compiled, not run on hardware
//
#include <hip/hip_runtime.h>
#include <hip/hip_bf16.h>

// ---------------------------------------------------------------------------
// Types for CDNA5 WMMA (gfx1250, wave32)
// ---------------------------------------------------------------------------
typedef __attribute__((ext_vector_type(16))) __bf16   v16bf;
typedef __attribute__((ext_vector_type(8)))  float    v8f;
typedef __attribute__((ext_vector_type(8)))  unsigned short ushort8;

union Frag {
    ushort8 h[2];   // two 16B global loads
    v16bf   v;      // WMMA operand view (same bits)
};

__device__ __forceinline__ unsigned short f2bf(float f) {
    union { float f; unsigned u; } v; v.f = f;
    unsigned r = v.u + 0x7FFFu + ((v.u >> 16) & 1u);   // round-to-nearest-even
    return (unsigned short)(r >> 16);
}
__device__ __forceinline__ float sigf(float x) { return 1.0f / (1.0f + expf(-x)); }

// Problem constants
#define VOCAB 32000
#define EMB   512
#define UNITS 1024
#define TDEC  64
#define BATCH 32
#define SENC  128
#define KLSTM (EMB + UNITS)      // 1536
#define NLSTM (4 * UNITS)        // 4096
#define KFC   (2 * UNITS)        // 2048

// ---------------------------------------------------------------------------
// Generic bf16 WMMA GEMM:  C[M,N] = A[M,K](row-major bf16) * Bt[N,K](bf16,
// i.e. B transposed) + bias[N], optional tanh epilogue. One 16x16 tile per
// wave, grid-stride over tiles.  Used for the M=4096 enc_proj precompute.
// ---------------------------------------------------------------------------
__global__ void k_gemm_bf16(const unsigned short* __restrict__ A,
                            const unsigned short* __restrict__ Bt,
                            float* __restrict__ C,
                            const float* __restrict__ bias,
                            int M, int N, int K, int act)
{
    const int lane   = threadIdx.x & 31;
    const int wid    = (blockIdx.x * blockDim.x + threadIdx.x) >> 5;
    const int nwaves = (gridDim.x * blockDim.x) >> 5;
    const int mt = M >> 4, nt = N >> 4, total = mt * nt;

    const int lsub = lane & 15;          // position within 16
    const int lhi  = lane >> 4;          // 0 or 1
    const int kh   = lhi << 3;           // A: K sub-offset 0 or 8
    const int kb16 = lhi << 3;           // B: K sub-offset 0 or 8

    for (int tile = wid; tile < total; tile += nwaves) {
        const int m0 = (tile % mt) << 4;
        const int n0 = (tile / mt) << 4;

        const unsigned short* arow = A  + (size_t)(m0 + lsub) * K;
        const unsigned short* brow = Bt + (size_t)(n0 + lsub) * K + kb16;

        v8f acc = {};
        for (int k0 = 0; k0 < K; k0 += 32) {
            Frag a, b;
            // A 16x32 bf16 fragment (ISA 7.12.2): elems 0..7 = K[kh..kh+7],
            // elems 8..15 = K[16+kh..16+kh+7]
            a.h[0] = *(const ushort8*)(arow + k0 + kh);
            a.h[1] = *(const ushort8*)(arow + k0 + 16 + kh);
            // B 32x16 fragment: lane covers N=n0+lsub, K = k0 + kb16 + [0..7], k0 + 16 + kb16 + [0..7]
            b.h[0] = *(const ushort8*)(brow + k0);
            b.h[1] = *(const ushort8*)(brow + k0 + 16);
            acc = __builtin_amdgcn_wmma_f32_16x16x32_bf16(
                      false, a.v, false, b.v, (short)0, acc, false, false);
        }

        const int col  = n0 + lsub;
        const int row0 = m0 + (lhi << 3);
        const float bb = bias ? bias[col] : 0.0f;
#pragma unroll
        for (int v = 0; v < 8; ++v) {
            float x = acc[v] + bb;
            if (act) x = tanhf(x);
            C[(size_t)(row0 + v) * N + col] = x;
        }
    }
}

// ---------------------------------------------------------------------------
// Specialized M=32 bf16 WMMA GEMM.  Each wave owns a full 32-row x 16-col
// strip: two f32 accumulators share every B fragment, so the dominant
// fc_W (Bt) stream is loaded ONCE per two v_wmma ops.  Grid-stride over the
// N/16 column strips.
// ---------------------------------------------------------------------------
__global__ void k_gemm_bf16_m32(const unsigned short* __restrict__ A,
                                const unsigned short* __restrict__ Bt,
                                float* __restrict__ C,
                                const float* __restrict__ bias,
                                int N, int K, int act)
{
    const int lane   = threadIdx.x & 31;
    const int wid    = (blockIdx.x * blockDim.x + threadIdx.x) >> 5;
    const int nwaves = (gridDim.x * blockDim.x) >> 5;
    const int strips = N >> 4;

    const int lsub = lane & 15;
    const int lhi  = lane >> 4;
    const int kh   = lhi << 3;           // A: K sub-offset 0 or 8
    const int kb16 = lhi << 3;           // B: K sub-offset 0 or 8

    const unsigned short* arow0 = A + (size_t)lsub * K;          // rows 0..15
    const unsigned short* arow1 = A + (size_t)(16 + lsub) * K;   // rows 16..31

    for (int s = wid; s < strips; s += nwaves) {
        const int n0 = s << 4;
        const unsigned short* brow = Bt + (size_t)(n0 + lsub) * K + kb16;

        v8f acc0 = {}, acc1 = {};
        for (int k0 = 0; k0 < K; k0 += 32) {
            Frag a0, a1, b;
            a0.h[0] = *(const ushort8*)(arow0 + k0 + kh);
            a0.h[1] = *(const ushort8*)(arow0 + k0 + 16 + kh);
            a1.h[0] = *(const ushort8*)(arow1 + k0 + kh);
            a1.h[1] = *(const ushort8*)(arow1 + k0 + 16 + kh);
            b.h[0]  = *(const ushort8*)(brow + k0);
            b.h[1]  = *(const ushort8*)(brow + k0 + 16);
            acc0 = __builtin_amdgcn_wmma_f32_16x16x32_bf16(
                       false, a0.v, false, b.v, (short)0, acc0, false, false);
            acc1 = __builtin_amdgcn_wmma_f32_16x16x32_bf16(
                       false, a1.v, false, b.v, (short)0, acc1, false, false);
        }

        const int col  = n0 + lsub;
        const int row0 = lhi << 3;       // rows 0..15 half
        const float bb = bias ? bias[col] : 0.0f;
#pragma unroll
        for (int v = 0; v < 8; ++v) {
            float x = acc0[v] + bb;
            if (act) x = tanhf(x);
            C[(size_t)(row0 + v) * N + col] = x;
        }
#pragma unroll
        for (int v = 0; v < 8; ++v) {
            float x = acc1[v] + bb;
            if (act) x = tanhf(x);
            C[(size_t)(16 + row0 + v) * N + col] = x;
        }
    }
}

// ---------------------------------------------------------------------------
// One-time conversion kernels
// ---------------------------------------------------------------------------
// outT[n*K + k] = bf16(in[k*N + n])   (fp32 [K,N] -> bf16 transposed [N,K])
__global__ void k_cvt_t(unsigned short* __restrict__ outT,
                        const float* __restrict__ in, int K, int N)
{
    size_t total = (size_t)K * N;
    size_t stride = (size_t)gridDim.x * blockDim.x;
    for (size_t i = (size_t)blockIdx.x * blockDim.x + threadIdx.x; i < total; i += stride) {
        size_t k = i / N, n = i - k * N;
        outT[n * (size_t)K + k] = f2bf(in[i]);
    }
}

// Concatenate + transpose LSTM weights: Wlt[n*1536 + k], k<512 from kernel,
// else from recurrent matrix. Both stored [K,4096] row-major fp32.
__global__ void k_lstm_cat_t(unsigned short* __restrict__ Wlt,
                             const float* __restrict__ lk,
                             const float* __restrict__ lr)
{
    size_t total = (size_t)NLSTM * KLSTM;
    size_t stride = (size_t)gridDim.x * blockDim.x;
    for (size_t i = (size_t)blockIdx.x * blockDim.x + threadIdx.x; i < total; i += stride) {
        size_t n = i / KLSTM, k = i - n * KLSTM;
        float v = (k < EMB) ? lk[k * NLSTM + n] : lr[(k - EMB) * NLSTM + n];
        Wlt[i] = f2bf(v);
    }
}

// Plain fp32 -> bf16 convert (same layout)
__global__ void k_cvt(unsigned short* __restrict__ out,
                      const float* __restrict__ in, size_t count)
{
    size_t stride = (size_t)gridDim.x * blockDim.x;
    for (size_t i = (size_t)blockIdx.x * blockDim.x + threadIdx.x; i < count; i += stride)
        out[i] = f2bf(in[i]);
}

// Init decoder state: h, c copies (must not mutate inputs), start tokens.
__global__ void k_init(float* __restrict__ h, float* __restrict__ c,
                       int* __restrict__ tok,
                       const float* __restrict__ eh, const float* __restrict__ ec)
{
    int i = blockIdx.x * blockDim.x + threadIdx.x;
    if (i < BATCH * UNITS) { h[i] = eh[i]; c[i] = ec[i]; }
    if (i < BATCH) tok[i] = 2;  // START_TOK
}

// ---------------------------------------------------------------------------
// Per-step kernels
// ---------------------------------------------------------------------------
// xh[b, 0:512] = bf16(emb[tok[b]]);  xh[b, 512:1536] = bf16(h[b])
__global__ void k_build_xh(unsigned short* __restrict__ xh,
                           const float* __restrict__ emb,
                           const int* __restrict__ tok,
                           const float* __restrict__ h)
{
    int i = blockIdx.x * blockDim.x + threadIdx.x;
    if (i >= BATCH * KLSTM) return;
    int b = i / KLSTM, k = i - b * KLSTM;
    float v = (k < EMB) ? emb[(size_t)tok[b] * EMB + k]
                        : h[(size_t)b * UNITS + (k - EMB)];
    xh[i] = f2bf(v);
}

// LSTM pointwise update from gates [B, 4U] (keras gate order i,f,g,o)
__global__ void k_lstm_update(float* __restrict__ h, float* __restrict__ c,
                              const float* __restrict__ gates)
{
    int i = blockIdx.x * blockDim.x + threadIdx.x;
    if (i >= BATCH * UNITS) return;
    int b = i >> 10, u = i & (UNITS - 1);
    const float* g = gates + (size_t)b * NLSTM;
    float ig = sigf(g[u]);
    float fg = sigf(g[UNITS + u]);
    float gg = tanhf(g[2 * UNITS + u]);
    float og = sigf(g[3 * UNITS + u]);
    float cn = fg * c[i] + ig * gg;
    c[i] = cn;
    h[i] = og * tanhf(cn);
}

// Attention: scores = h . enc_proj, softmax over S, ctx = attn @ enc_outputs,
// then pack ch[b] = [bf16(ctx) | bf16(h)].  One block per batch row, 128 thr.
__global__ void k_attn(unsigned short* __restrict__ ch,
                       const float* __restrict__ h,
                       const float* __restrict__ encProj,
                       const float* __restrict__ encO)
{
    const int b = blockIdx.x, tid = threadIdx.x;  // 128 threads
    __shared__ float sh[UNITS];
    __shared__ float red[SENC];
    __shared__ float attn[SENC];

    for (int u = tid; u < UNITS; u += SENC) sh[u] = h[(size_t)b * UNITS + u];
    __syncthreads();

    const float* ep = encProj + ((size_t)b * SENC + tid) * UNITS;
    float sc = 0.0f;
    for (int u = 0; u < UNITS; ++u) sc += sh[u] * ep[u];

    red[tid] = sc; __syncthreads();
    for (int o = SENC / 2; o > 0; o >>= 1) {
        if (tid < o) red[tid] = fmaxf(red[tid], red[tid + o]);
        __syncthreads();
    }
    const float mx = red[0]; __syncthreads();
    float e = expf(sc - mx);
    red[tid] = e; __syncthreads();
    for (int o = SENC / 2; o > 0; o >>= 1) {
        if (tid < o) red[tid] += red[tid + o];
        __syncthreads();
    }
    const float inv = 1.0f / red[0]; __syncthreads();
    attn[tid] = e * inv; __syncthreads();

    for (int u = tid; u < UNITS; u += SENC) {
        float acc = 0.0f;
        const float* eo = encO + (size_t)b * SENC * UNITS + u;
        for (int s = 0; s < SENC; ++s) acc += attn[s] * eo[(size_t)s * UNITS];
        ch[(size_t)b * KFC + u]         = f2bf(acc);
        ch[(size_t)b * KFC + UNITS + u] = f2bf(sh[u]);
    }
}

// Vocab reduction: logits already tanh'd.  argmax (first-max tiebreak),
// logsumexp, write logp row into d_out, greedy token feedback + pred output.
__global__ void k_vocab(float* __restrict__ dout,
                        const float* __restrict__ logits,
                        int* __restrict__ tok, int t)
{
    const int b = blockIdx.x, tid = threadIdx.x;  // 256 threads
    __shared__ float bv[256];
    __shared__ int   bi[256];
    __shared__ float sv[256];

    const float* row = logits + (size_t)b * VOCAB;
    float vmax = -3.4e38f; int vidx = 0;
    for (int n = tid; n < VOCAB; n += 256) {
        float v = row[n];
        if (v > vmax) { vmax = v; vidx = n; }
    }
    bv[tid] = vmax; bi[tid] = vidx; __syncthreads();
    for (int o = 128; o > 0; o >>= 1) {
        if (tid < o) {
            if (bv[tid + o] > bv[tid] ||
                (bv[tid + o] == bv[tid] && bi[tid + o] < bi[tid])) {
                bv[tid] = bv[tid + o]; bi[tid] = bi[tid + o];
            }
        }
        __syncthreads();
    }
    const float gmax = bv[0]; const int gidx = bi[0]; __syncthreads();

    float s = 0.0f;
    for (int n = tid; n < VOCAB; n += 256) s += expf(row[n] - gmax);
    sv[tid] = s; __syncthreads();
    for (int o = 128; o > 0; o >>= 1) {
        if (tid < o) sv[tid] += sv[tid + o];
        __syncthreads();
    }
    const float lse = logf(sv[0]) + gmax;

    float* out = dout + (size_t)BATCH * TDEC
                      + ((size_t)b * TDEC + t) * VOCAB;
    for (int n = tid; n < VOCAB; n += 256) out[n] = row[n] - lse;

    if (tid == 0) {
        tok[b] = gidx;
        dout[(size_t)b * TDEC + t] = (float)gidx;   // pred_output
    }
}

// ---------------------------------------------------------------------------
// Host-side orchestration
// ---------------------------------------------------------------------------
extern "C" void kernel_launch(void* const* d_in, const int* in_sizes, int n_in,
                              void* d_out, int out_size, void* d_ws, size_t ws_size,
                              hipStream_t stream)
{
    (void)in_sizes; (void)n_in; (void)out_size; (void)ws_size;

    const float* enc_h   = (const float*)d_in[0];
    const float* enc_c   = (const float*)d_in[1];
    // d_in[2] = target (unused by the reference decode path)
    const float* enc_out = (const float*)d_in[3];
    const float* emb     = (const float*)d_in[4];
    const float* lk      = (const float*)d_in[5];
    const float* lr      = (const float*)d_in[6];
    const float* lbias   = (const float*)d_in[7];
    const float* Wa      = (const float*)d_in[8];
    const float* Wab     = (const float*)d_in[9];
    const float* fcW     = (const float*)d_in[10];
    const float* fcb     = (const float*)d_in[11];
    float* out = (float*)d_out;

    // Workspace carve-up (256B aligned)
    char* ws = (char*)d_ws;
    size_t off = 0;
    auto alloc = [&](size_t bytes) -> char* {
        char* p = ws + off;
        off += (bytes + 255) & ~(size_t)255;
        return p;
    };
    unsigned short* fcWt  = (unsigned short*)alloc((size_t)VOCAB * KFC * 2);       // 131 MB, L2-resident after step 0
    unsigned short* Wlt   = (unsigned short*)alloc((size_t)NLSTM * KLSTM * 2);     // 12.6 MB
    unsigned short* Wat   = (unsigned short*)alloc((size_t)UNITS * UNITS * 2);     // 2 MB
    unsigned short* encBf = (unsigned short*)alloc((size_t)BATCH * SENC * UNITS * 2);
    float*          encPj = (float*)alloc((size_t)BATCH * SENC * UNITS * 4);       // 16.8 MB
    float*          h     = (float*)alloc((size_t)BATCH * UNITS * 4);
    float*          c     = (float*)alloc((size_t)BATCH * UNITS * 4);
    float*          gates = (float*)alloc((size_t)BATCH * NLSTM * 4);
    unsigned short* xh    = (unsigned short*)alloc((size_t)BATCH * KLSTM * 2);
    unsigned short* ch    = (unsigned short*)alloc((size_t)BATCH * KFC * 2);
    float*          logit = (float*)alloc((size_t)BATCH * VOCAB * 4);
    int*            tok   = (int*)alloc((size_t)BATCH * 4);

    // --- one-time precompute -------------------------------------------------
    k_init<<<(BATCH * UNITS + 255) / 256, 256, 0, stream>>>(h, c, tok, enc_h, enc_c);
    k_cvt_t<<<8192, 256, 0, stream>>>(fcWt, fcW, KFC, VOCAB);          // fc_W -> bf16 [N,K]
    k_cvt_t<<<1024, 256, 0, stream>>>(Wat, Wa, UNITS, UNITS);          // Wa   -> bf16 [N,K]
    k_lstm_cat_t<<<4096, 256, 0, stream>>>(Wlt, lk, lr);               // [W_k;W_r] -> bf16 [N,K]
    k_cvt<<<4096, 256, 0, stream>>>(encBf, enc_out, (size_t)BATCH * SENC * UNITS);

    // enc_proj = enc_out @ Wa + Wa_b :  M=4096, N=1024, K=1024 (WMMA)
    k_gemm_bf16<<<2048, 256, 0, stream>>>(encBf, Wat, encPj, Wab,
                                          BATCH * SENC, UNITS, UNITS, /*act=*/0);

    // --- sequential greedy decode -------------------------------------------
    for (int t = 0; t < TDEC; ++t) {
        k_build_xh<<<(BATCH * KLSTM + 255) / 256, 256, 0, stream>>>(xh, emb, tok, h);

        // gates = xh @ [W_k;W_r] + bias :  M=32, N=4096, K=1536 (WMMA, 256 strips)
        k_gemm_bf16_m32<<<32, 256, 0, stream>>>(xh, Wlt, gates, lbias,
                                                NLSTM, KLSTM, /*act=*/0);

        k_lstm_update<<<(BATCH * UNITS + 255) / 256, 256, 0, stream>>>(h, c, gates);

        k_attn<<<BATCH, SENC, 0, stream>>>(ch, h, encPj, enc_out);

        // logits = tanh([ctx|h] @ fc_W + fc_b) : M=32, N=32000 (2000 strips, WMMA)
        k_gemm_bf16_m32<<<250, 256, 0, stream>>>(ch, fcWt, logit, fcb,
                                                 VOCAB, KFC, /*act=*/1);

        k_vocab<<<BATCH, 256, 0, stream>>>(out, logit, tok, t);
    }
}
